// AAttn_60000693125585
// MI455X (gfx1250) — matrix operand order repaired and checked by reading, not compiled, not run on hardware
//
#include <hip/hip_runtime.h>

typedef __bf16 bf16_t;
typedef __attribute__((ext_vector_type(16))) __bf16 v16bf;
typedef __attribute__((ext_vector_type(8)))  float  v8f;
typedef __attribute__((ext_vector_type(4)))  unsigned int v4u;
typedef __attribute__((ext_vector_type(8)))  int  v8i;
typedef __attribute__((ext_vector_type(4)))  int  v4i;

// ---------------------------------------------------------------------------
// TDM: load a contiguous 16x256 bf16 strip (8KB) from global to LDS.
// D# group0: count=1, lds_addr, global_addr[56:0], type=2 ("image").
// D# group1: data_size=2B, tensor_dim0=256, tensor_dim1=16, tile_dim0=256,
//            tile_dim1=16, tensor_dim0_stride=256.  Groups 2/3 unused (2D).
// ---------------------------------------------------------------------------
__device__ inline void tdm_load_strip_16x256(const bf16_t* gptr, unsigned lds_off)
{
    unsigned long long ga = (unsigned long long)(uintptr_t)gptr;
    v4u g0;
    g0[0] = 1u;                                        // count=1, user mode
    g0[1] = lds_off;                                   // LDS byte address
    g0[2] = (unsigned)ga;                              // global_addr[31:0]
    g0[3] = (unsigned)((ga >> 32) & 0x1FFFFFFu) | (2u << 30);  // [56:32] | type=2
    v8i g1;
    g1[0] = (int)(1u << 16);                           // data_size=1 (2 bytes)
    g1[1] = (int)(256u << 16);                         // tensor_dim0 = 256
    g1[2] = (int)(16u  << 16);                         // tensor_dim1 = 16
    g1[3] = (int)(256u << 16);                         // tile_dim0 = 256
    g1[4] = 16;                                        // tile_dim1 = 16
    g1[5] = 256;                                       // tensor_dim0_stride = 256
    g1[6] = 0;
    g1[7] = 0;
    v4i z4 = {0, 0, 0, 0};
#if defined(__clang_major__) && (__clang_major__ >= 23)
    v8i z8 = {0, 0, 0, 0, 0, 0, 0, 0};
    __builtin_amdgcn_tensor_load_to_lds(g0, g1, z4, z4, z8, 0);
#else
    __builtin_amdgcn_tensor_load_to_lds(g0, g1, z4, z4, 0);
#endif
}

// ---------------------------------------------------------------------------
// Kernel 0: fold BN into weights. qkv/proj weights -> bf16, pe weights -> f32.
// ---------------------------------------------------------------------------
__global__ void __launch_bounds__(256) prep_kernel(
    const float* __restrict__ w_qkv, const float* __restrict__ g_qkv,
    const float* __restrict__ b_qkv, const float* __restrict__ m_qkv,
    const float* __restrict__ v_qkv,
    const float* __restrict__ w_pe,  const float* __restrict__ g_pe,
    const float* __restrict__ b_pe,  const float* __restrict__ m_pe,
    const float* __restrict__ v_pe,
    const float* __restrict__ w_proj,const float* __restrict__ g_proj,
    const float* __restrict__ b_proj,const float* __restrict__ m_proj,
    const float* __restrict__ v_proj,
    bf16_t* __restrict__ Wq, float* __restrict__ biasq,
    bf16_t* __restrict__ Wp, float* __restrict__ biasp,
    float*  __restrict__ Wpe, float* __restrict__ biaspe)
{
    int idx = blockIdx.x * 256 + threadIdx.x;              // 768*256 threads
    if (idx < 768 * 256) {
        int o = idx >> 8;
        float s = g_qkv[o] * rsqrtf(v_qkv[o] + 1e-5f);
        Wq[idx] = (bf16_t)(w_qkv[idx] * s);
        if ((idx & 255) == 0) biasq[o] = b_qkv[o] - m_qkv[o] * s;
    }
    if (idx < 256 * 256) {
        int o = idx >> 8;
        float s = g_proj[o] * rsqrtf(v_proj[o] + 1e-5f);
        Wp[idx] = (bf16_t)(w_proj[idx] * s);
        if ((idx & 255) == 0) biasp[o] = b_proj[o] - m_proj[o] * s;
    }
    if (idx < 256 * 49) {
        int c = idx / 49;
        float s = g_pe[c] * rsqrtf(v_pe[c] + 1e-5f);
        Wpe[idx] = w_pe[idx] * s;
        if (idx - c * 49 == 0) biaspe[c] = b_pe[c] - m_pe[c] * s;
    }
}

// ---------------------------------------------------------------------------
// Kernel 1: x NCHW f32  ->  Xb[row = b*4096+hw][c] bf16
// ---------------------------------------------------------------------------
__global__ void __launch_bounds__(256) pack_x_kernel(
    const float* __restrict__ x, bf16_t* __restrict__ Xb)
{
    int idx = blockIdx.x * 256 + threadIdx.x;              // 2*4096*256
    int c  = idx & 255;
    int hw = (idx >> 8) & 4095;
    int b  = idx >> 20;
    Xb[idx] = (bf16_t)x[((size_t)b * 256 + c) * 4096 + hw];
}

// ---------------------------------------------------------------------------
// Kernel 2: qkv GEMM  Y[8192,768] = Xb * Wq^T + bias, scatter to Q/K/V + v2.
// Block = 16-row strip x 128 cols (8 waves, one 16x16 tile each).
// A strip staged via TDM tensor_load_to_lds; B streamed from global.
// ---------------------------------------------------------------------------
__global__ void __launch_bounds__(256) gemm_qkv_kernel(
    const bf16_t* __restrict__ Xb, const bf16_t* __restrict__ Wq,
    const float*  __restrict__ bias,
    bf16_t* __restrict__ Qb, bf16_t* __restrict__ Kb, bf16_t* __restrict__ Vb,
    float*  __restrict__ v2)
{
    __shared__ bf16_t lds_A[16 * 256];                     // 8KB
    const int lane = threadIdx.x & 31;
    const int wave = threadIdx.x >> 5;
    int tr  = blockIdx.x / 6;                              // 512 row strips
    int tcg = blockIdx.x - tr * 6;                         // 6 col groups
    int tc  = tcg * 8 + wave;                              // 48 col tiles
    const int rowb = tr * 16, colb = tc * 16;
    const int m  = lane & 15;
    const int gs = (lane >> 4) << 3;

    if (wave == 0) {
        tdm_load_strip_16x256(Xb + (size_t)rowb * 256,
                              (unsigned)(uintptr_t)&lds_A[0]);
        __builtin_amdgcn_s_wait_tensorcnt(0);
    }
    __syncthreads();

    v8f acc = {};
#pragma unroll
    for (int kk = 0; kk < 256; kk += 32) {
        v16bf a, b;
#pragma unroll
        for (int e = 0; e < 16; ++e) {
            int r = e >> 1, hi = e & 1;
            int kA = ((r >> 2) << 4) + gs + ((r & 3) << 1) + hi;   // A 16x32 layout
            a[e] = lds_A[m * 256 + kk + kA];
            int kB = ((e >> 3) << 4) + gs + (e & 7);                // B 32x16 layout
            b[e] = Wq[(size_t)(colb + m) * 256 + kk + kB];
        }
        acc = __builtin_amdgcn_wmma_f32_16x16x32_bf16(
                  false, a, false, b, (short)0, acc, false, false);
    }
    const int g   = lane >> 4;
    const int col = colb + m;
    const float bi = bias[col];
    const int head = col / 96, j = col - head * 96;
#pragma unroll
    for (int r = 0; r < 8; ++r) {
        int row = rowb + r + 8 * g;
        float val = acc[r] + bi;
        int bimg = row >> 12;
        int n    = row & 4095;
        int ba   = bimg * 4 + (n >> 10);
        int na   = n & 1023;
        size_t qidx = ((size_t)(ba * 8 + head) * 1024 + na) * 32;
        if (j < 32) {
            Qb[qidx + j] = (bf16_t)val;
        } else if (j < 64) {
            Kb[qidx + (j - 32)] = (bf16_t)val;
        } else {
            int d = j - 64;
            Vb[qidx + d] = (bf16_t)val;
            v2[((size_t)bimg * 256 + (head * 32 + d)) * 4096 + n] = val;
        }
    }
}

// ---------------------------------------------------------------------------
// Kernel 3: flash attention. One wave = 16 query rows of one (ba,head),
// streaming K/V in chunks of 32, P routed through LDS for layout conversion.
// ---------------------------------------------------------------------------
__global__ void __launch_bounds__(256) attn_kernel(
    const bf16_t* __restrict__ Qb, const bf16_t* __restrict__ Kb,
    const bf16_t* __restrict__ Vb, float* __restrict__ Obuf)
{
    __shared__ bf16_t pstage[8 * 16 * 32];                 // 1KB per wave
    const int lane = threadIdx.x & 31;
    const int wave = threadIdx.x >> 5;
    int s = blockIdx.x * 8 + wave;                          // 4096 strips
    int bh = s >> 6;
    int rowTile = s & 63;
    const bf16_t* Q = Qb + (size_t)bh * 1024 * 32;
    const bf16_t* K = Kb + (size_t)bh * 1024 * 32;
    const bf16_t* V = Vb + (size_t)bh * 1024 * 32;
    const int m_ = lane & 15;
    const int g  = lane >> 4;
    const int gs = g << 3;

    v16bf aq;
#pragma unroll
    for (int e = 0; e < 16; ++e) {
        int r = e >> 1, hi = e & 1;
        int kA = ((r >> 2) << 4) + gs + ((r & 3) << 1) + hi;
        aq[e] = Q[(size_t)(rowTile * 16 + m_) * 32 + kA];
    }

    float mi[8], li[8];
#pragma unroll
    for (int r = 0; r < 8; ++r) { mi[r] = -1e30f; li[r] = 0.0f; }
    v8f o0 = {}, o1 = {};
    bf16_t* pl = pstage + wave * 512;
    const float scl = 0.17677669529663687f;                // 1/sqrt(32)

    for (int mc = 0; mc < 1024; mc += 32) {
        v16bf bk0, bk1;
#pragma unroll
        for (int e = 0; e < 16; ++e) {
            int kB = ((e >> 3) << 4) + gs + (e & 7);        // d index
            bk0[e] = K[(size_t)(mc + m_) * 32 + kB];
            bk1[e] = K[(size_t)(mc + 16 + m_) * 32 + kB];
        }
        v8f z = {};
        v8f s0 = __builtin_amdgcn_wmma_f32_16x16x32_bf16(
                     false, aq, false, bk0, (short)0, z, false, false);
        v8f s1 = __builtin_amdgcn_wmma_f32_16x16x32_bf16(
                     false, aq, false, bk1, (short)0, z, false, false);
#pragma unroll
        for (int r = 0; r < 8; ++r) {
            float x0 = s0[r] * scl, x1 = s1[r] * scl;
            float rmax = fmaxf(x0, x1);
#pragma unroll
            for (int msk = 1; msk < 16; msk <<= 1)
                rmax = fmaxf(rmax, __shfl_xor(rmax, msk, 32));
            float nm = fmaxf(mi[r], rmax);
            float p0 = __expf(x0 - nm), p1 = __expf(x1 - nm);
            float rsum = p0 + p1;
#pragma unroll
            for (int msk = 1; msk < 16; msk <<= 1)
                rsum += __shfl_xor(rsum, msk, 32);
            float corr = __expf(mi[r] - nm);
            li[r] = li[r] * corr + rsum;
            mi[r] = nm;
            o0[r] *= corr; o1[r] *= corr;
            int prow = r + 8 * g;
            pl[prow * 32 + m_]      = (bf16_t)p0;
            pl[prow * 32 + 16 + m_] = (bf16_t)p1;
        }
        __syncthreads();                                    // uniform: 32 iters for all waves
        v16bf ap, bv0, bv1;
#pragma unroll
        for (int e = 0; e < 16; ++e) {
            int r = e >> 1, hi = e & 1;
            int kA = ((r >> 2) << 4) + gs + ((r & 3) << 1) + hi;
            ap[e] = pl[m_ * 32 + kA];
            int kB = ((e >> 3) << 4) + gs + (e & 7);        // m-local index
            bv0[e] = V[(size_t)(mc + kB) * 32 + m_];
            bv1[e] = V[(size_t)(mc + kB) * 32 + 16 + m_];
        }
        o0 = __builtin_amdgcn_wmma_f32_16x16x32_bf16(
                 false, ap, false, bv0, (short)0, o0, false, false);
        o1 = __builtin_amdgcn_wmma_f32_16x16x32_bf16(
                 false, ap, false, bv1, (short)0, o1, false, false);
        __syncthreads();
    }

    int ba = bh >> 3, head = bh & 7;
    int b = ba >> 2, area = ba & 3;
#pragma unroll
    for (int r = 0; r < 8; ++r) {
        float inv = 1.0f / li[r];
        int na = rowTile * 16 + r + 8 * g;
        size_t row = (size_t)b * 4096 + area * 1024 + na;
        int c0 = head * 32 + m_;
        Obuf[row * 256 + c0]      = o0[r] * inv;
        Obuf[row * 256 + c0 + 16] = o1[r] * inv;
    }
}

// ---------------------------------------------------------------------------
// Kernel 4: pe = dwconv7(v2)*BN ; F = bf16(Obuf + pe)   (F row-major [8192,256])
// ---------------------------------------------------------------------------
__global__ void __launch_bounds__(256) pe_fuse_kernel(
    const float* __restrict__ v2, const float* __restrict__ wpe,
    const float* __restrict__ bpe, const float* __restrict__ Obuf,
    bf16_t* __restrict__ F)
{
    int idx = blockIdx.x * 256 + threadIdx.x;              // 2*256*4096
    int hw = idx & 4095;
    int c  = (idx >> 12) & 255;
    int b  = idx >> 20;
    int h = hw >> 6, w = hw & 63;
    float acc = bpe[c];
    const float* vbase = v2 + ((size_t)b * 256 + c) * 4096;
    const float* wb = wpe + c * 49;
#pragma unroll
    for (int dh = 0; dh < 7; ++dh) {
        int hh = h + dh - 3;
        if (hh < 0 || hh > 63) continue;
#pragma unroll
        for (int dw = 0; dw < 7; ++dw) {
            int ww = w + dw - 3;
            if (ww < 0 || ww > 63) continue;
            acc += vbase[hh * 64 + ww] * wb[dh * 7 + dw];
        }
    }
    size_t row = (size_t)b * 4096 + hw;
    F[row * 256 + c] = (bf16_t)(Obuf[row * 256 + c] + acc);
}

// ---------------------------------------------------------------------------
// Kernel 5: proj GEMM  out = F * Wp^T + bias  -> NCHW f32.
// Same TDM-staged A-strip structure as kernel 2.
// ---------------------------------------------------------------------------
__global__ void __launch_bounds__(256) gemm_proj_kernel(
    const bf16_t* __restrict__ F, const bf16_t* __restrict__ Wp,
    const float* __restrict__ bias, float* __restrict__ out)
{
    __shared__ bf16_t lds_A[16 * 256];                     // 8KB
    const int lane = threadIdx.x & 31;
    const int wave = threadIdx.x >> 5;
    int tr  = blockIdx.x >> 1;                             // 512 row strips
    int tcg = blockIdx.x & 1;                              // 2 col groups
    int tc  = tcg * 8 + wave;                              // 16 col tiles
    const int rowb = tr * 16, colb = tc * 16;
    const int m  = lane & 15;
    const int gs = (lane >> 4) << 3;

    if (wave == 0) {
        tdm_load_strip_16x256(F + (size_t)rowb * 256,
                              (unsigned)(uintptr_t)&lds_A[0]);
        __builtin_amdgcn_s_wait_tensorcnt(0);
    }
    __syncthreads();

    v8f acc = {};
#pragma unroll
    for (int kk = 0; kk < 256; kk += 32) {
        v16bf a, b;
#pragma unroll
        for (int e = 0; e < 16; ++e) {
            int r = e >> 1, hi = e & 1;
            int kA = ((r >> 2) << 4) + gs + ((r & 3) << 1) + hi;
            a[e] = lds_A[m * 256 + kk + kA];
            int kB = ((e >> 3) << 4) + gs + (e & 7);
            b[e] = Wp[(size_t)(colb + m) * 256 + kk + kB];
        }
        acc = __builtin_amdgcn_wmma_f32_16x16x32_bf16(
                  false, a, false, b, (short)0, acc, false, false);
    }
    const int g = lane >> 4;
    const int col = colb + m;
    const float bi = bias[col];
#pragma unroll
    for (int r = 0; r < 8; ++r) {
        int row = rowb + r + 8 * g;
        int b_ = row >> 12;
        int hw = row & 4095;
        out[((size_t)b_ * 256 + col) * 4096 + hw] = acc[r] + bi;
    }
}

// ---------------------------------------------------------------------------
// Launcher
// ---------------------------------------------------------------------------
extern "C" void kernel_launch(void* const* d_in, const int* in_sizes, int n_in,
                              void* d_out, int out_size, void* d_ws, size_t ws_size,
                              hipStream_t stream) {
    const float* x      = (const float*)d_in[0];
    const float* w_qkv  = (const float*)d_in[1];
    const float* g_qkv  = (const float*)d_in[2];
    const float* b_qkv  = (const float*)d_in[3];
    const float* m_qkv  = (const float*)d_in[4];
    const float* v_qkv  = (const float*)d_in[5];
    const float* w_pe   = (const float*)d_in[6];
    const float* g_pe   = (const float*)d_in[7];
    const float* b_pe   = (const float*)d_in[8];
    const float* m_pe   = (const float*)d_in[9];
    const float* v_pe   = (const float*)d_in[10];
    const float* w_proj = (const float*)d_in[11];
    const float* g_proj = (const float*)d_in[12];
    const float* b_proj = (const float*)d_in[13];
    const float* m_proj = (const float*)d_in[14];
    const float* v_proj = (const float*)d_in[15];
    float* out = (float*)d_out;

    char* ws = (char*)d_ws;
    const size_t XB  = 0;                      // 8192*256*2 bytes (reused as F)
    const size_t WQ  = XB  + 4194304;
    const size_t BQ  = WQ  + 393216;
    const size_t WP  = BQ  + 3072;
    const size_t BP  = WP  + 131072;
    const size_t WPE = BP  + 1024;
    const size_t BPE = WPE + 50176;
    const size_t QB  = BPE + 1024;
    const size_t KBo = QB  + 4194304;
    const size_t VB  = KBo + 4194304;
    const size_t V2  = VB  + 4194304;
    const size_t OB  = V2  + 8388608;

    bf16_t* Xb    = (bf16_t*)(ws + XB);
    bf16_t* Wq    = (bf16_t*)(ws + WQ);
    float*  biasq = (float*) (ws + BQ);
    bf16_t* Wp    = (bf16_t*)(ws + WP);
    float*  biasp = (float*) (ws + BP);
    float*  Wpe   = (float*) (ws + WPE);
    float*  biaspe= (float*) (ws + BPE);
    bf16_t* Qb    = (bf16_t*)(ws + QB);
    bf16_t* Kb    = (bf16_t*)(ws + KBo);
    bf16_t* Vb    = (bf16_t*)(ws + VB);
    float*  v2    = (float*) (ws + V2);
    float*  Obuf  = (float*) (ws + OB);
    bf16_t* F     = (bf16_t*)(ws + XB);        // reuse Xb region

    prep_kernel<<<768, 256, 0, stream>>>(
        w_qkv, g_qkv, b_qkv, m_qkv, v_qkv,
        w_pe,  g_pe,  b_pe,  m_pe,  v_pe,
        w_proj, g_proj, b_proj, m_proj, v_proj,
        Wq, biasq, Wp, biasp, Wpe, biaspe);

    pack_x_kernel<<<8192, 256, 0, stream>>>(x, Xb);

    gemm_qkv_kernel<<<3072, 256, 0, stream>>>(Xb, Wq, biasq, Qb, Kb, Vb, v2);

    attn_kernel<<<512, 256, 0, stream>>>(Qb, Kb, Vb, Obuf);

    pe_fuse_kernel<<<8192, 256, 0, stream>>>(v2, Wpe, biaspe, Obuf, F);

    gemm_proj_kernel<<<1024, 256, 0, stream>>>(F, Wp, biasp, out);
}
